// SigmoidAttention_72662256714429
// MI455X (gfx1250) — matrix operand order repaired and checked by reading, not compiled, not run on hardware
//
#include <hip/hip_runtime.h>
#include <stdint.h>

// ---------------------------------------------------------------------------
// Types for CDNA5 WMMA (wave32): v_wmma_f32_16x16x32_bf16
// ---------------------------------------------------------------------------
typedef __attribute__((ext_vector_type(16))) __bf16 v16bf;
typedef __attribute__((ext_vector_type(8)))  float  v8f;
typedef __attribute__((ext_vector_type(4)))  unsigned int u32x4;
typedef __attribute__((ext_vector_type(8)))  int i32x8;
typedef __attribute__((ext_vector_type(4)))  int i32x4;

union Frag { uint4 q[2]; v16bf v; };      // 32 bytes: one A or B operand
union AccU { v8f v; float f[8]; };        // 32 bytes: one C/D accumulator

__device__ __forceinline__ uint32_t pk_bf16(float a, float b) {
    union { float f; uint32_t u; } x, y;
    x.f = a; y.f = b;
    uint32_t lo = (x.u + 0x7FFFu + ((x.u >> 16) & 1u)) >> 16;
    uint32_t hi = (y.u + 0x7FFFu + ((y.u >> 16) & 1u)) >> 16;
    return lo | (hi << 16);
}
__device__ __forceinline__ uint16_t bf16u(float a) {
    union { float f; uint32_t u; } x; x.f = a;
    return (uint16_t)((x.u + 0x7FFFu + ((x.u >> 16) & 1u)) >> 16);
}

#define WMMA_BF16(a, b, c) \
    __builtin_amdgcn_wmma_f32_16x16x32_bf16(false, (a), false, (b), (short)0, (c), false, false)

// Problem constants
#define BATCH 4
#define SEQ   4096
#define DIM   1024
#define MTOT  (BATCH * SEQ)   // 16384

// ---------------------------------------------------------------------------
// LDS byte-offset of a __shared__ object (addrspace(3) pointers are the
// 32-bit LDS offset on AMDGPU).
// ---------------------------------------------------------------------------
__device__ __forceinline__ uint32_t lds_offset_of(const void* p) {
    return (uint32_t)(uintptr_t)(__attribute__((address_space(3))) const void*)p;
}

// ---------------------------------------------------------------------------
// Tensor Data Mover: async 2D bf16 tile load global->LDS with LDS padding of
// 16B after every 64B row (rebuilds the 80B bank-conflict-free row stride).
// D# packing per CDNA5 ISA.
// ---------------------------------------------------------------------------
__device__ __forceinline__ void tdm_load_2d_bf16(
    uint32_t lds_off, const uint16_t* gsrc,
    uint32_t tile_d0, uint32_t tile_d1,
    uint32_t tensor_d0, uint32_t tensor_d1, uint32_t stride0)
{
    uint64_t ga = (uint64_t)(uintptr_t)gsrc;
    u32x4 g0;
    g0.x = 1u;                                   // count=1, user descriptor
    g0.y = lds_off;                              // lds_addr [63:32]
    g0.z = (uint32_t)ga;                         // global_addr [95:64]
    g0.w = (uint32_t)(ga >> 32) | 0x80000000u;   // addr hi; type=2 -> bits[127:126]=10b
    i32x8 g1;
    // data_size=1 (2B), pad_enable, pad_interval=3 (every 16 DW = 64B),
    // pad_amount=3 (4 DW = 16B)
    g1[0] = (int)((1u << 16) | (1u << 20) | (3u << 22) | (3u << 25));
    g1[1] = (int)((tensor_d0 & 0xFFFFu) << 16);                    // tensor_dim0 lo
    g1[2] = (int)((tensor_d0 >> 16) | ((tensor_d1 & 0xFFFFu) << 16));
    g1[3] = (int)((tensor_d1 >> 16) | (tile_d0 << 16));            // tile_dim0
    g1[4] = (int)tile_d1;                                          // tile_dim1 (tile_dim2=0)
    g1[5] = (int)stride0;                                          // tensor_dim0_stride lo
    g1[6] = 0;
    g1[7] = 0;
    i32x4 z4 = {0, 0, 0, 0};
#if __has_include(<hip/amd_detail/amd_gfx1250_TDM.h>)
    i32x8 z8 = {0, 0, 0, 0, 0, 0, 0, 0};
    __builtin_amdgcn_tensor_load_to_lds(g0, g1, z4, z4, z8, 0);    // 6-arg toolchain
#else
    __builtin_amdgcn_tensor_load_to_lds(g0, g1, z4, z4, 0);        // 5-arg ROCm 7.2
#endif
}

// ---------------------------------------------------------------------------
// Kernel 0a: X fp32 -> bf16 (done once instead of 3x inside the GEMM)
// ---------------------------------------------------------------------------
__global__ __launch_bounds__(256) void cvt_x(const float* __restrict__ X,
                                             uint16_t* __restrict__ Xb)
{
    size_t i = ((size_t)blockIdx.x * 256 + threadIdx.x) * 8;
    float4 a = *(const float4*)(X + i);
    float4 b = *(const float4*)(X + i + 4);
    uint4 p;
    p.x = pk_bf16(a.x, a.y); p.y = pk_bf16(a.z, a.w);
    p.z = pk_bf16(b.x, b.y); p.w = pk_bf16(b.z, b.w);
    *(uint4*)(Xb + i) = p;
}

// ---------------------------------------------------------------------------
// Kernel 0b: W fp32 [K][N] -> Wt bf16 [z][N][K] (transposed via LDS tile so
// the GEMM's B staging becomes a contiguous-row 2D tile = TDM friendly).
// ---------------------------------------------------------------------------
__global__ __launch_bounds__(256) void cvt_w(
    const float* __restrict__ Wq, const float* __restrict__ Wk,
    const float* __restrict__ Wv, uint16_t* __restrict__ Wt)
{
    const int z = blockIdx.z;
    const float* W = (z == 0) ? Wq : (z == 1) ? Wk : Wv;
    __shared__ float t[32][33];
    const int kb = blockIdx.x * 32, nb = blockIdx.y * 32;
    const int tx = threadIdx.x & 31, ty = threadIdx.x >> 5;   // 32 x 8
#pragma unroll
    for (int j = 0; j < 32; j += 8)
        t[ty + j][tx] = W[(size_t)(kb + ty + j) * DIM + nb + tx];
    __syncthreads();
#pragma unroll
    for (int j = 0; j < 32; j += 8)
        Wt[((size_t)z * DIM + nb + ty + j) * DIM + kb + tx] = bf16u(t[tx][ty + j]);
}

// ---------------------------------------------------------------------------
// Kernel 1: P = Xb @ Wt^T + b (bf16 in, fp32 accum, bf16 out).
// grid.z selects Q/K/V. Q,K row-major [MTOT][DIM]; V transposed [DIM][MTOT].
// Block tile 128x64, K-step 32; tiles staged by TDM, double-buffered on
// TENSORcnt; 8 waves each own a 32x32 region (2x2 WMMA tiles).
// ---------------------------------------------------------------------------
__global__ __launch_bounds__(256) void qkv_gemm(
    const uint16_t* __restrict__ Xb, const uint16_t* __restrict__ Wt,
    const float* __restrict__ bq, const float* __restrict__ bk,
    const float* __restrict__ bv,
    uint16_t* __restrict__ Qb, uint16_t* __restrict__ Kb,
    uint16_t* __restrict__ Vt)
{
    const int z = blockIdx.z;
    const float* bias = (z == 0) ? bq : (z == 1) ? bk : bv;
    uint16_t* outp    = (z == 0) ? Qb : (z == 1) ? Kb : Vt;
    const bool transposed = (z == 2);

    // 40-short rows (80B): produced by TDM pad fields, conflict-free reads
    __shared__ uint16_t As[2][128][40];
    __shared__ uint16_t Bs[2][64][40];

    const int tid  = threadIdx.x;
    const int wave = tid >> 5;
    const int lane = tid & 31;
    const int mb   = blockIdx.y * 128;
    const int nb   = blockIdx.x * 64;
    const int wm   = (wave & 3) * 32;
    const int wn   = (wave >> 2) * 32;

    const int l15  = lane & 15;
    const int aoff = (lane < 16) ? 0 : 16;   // A-frag K base
    const int boff = (lane < 16) ? 0 : 16;   // B-frag K base

    AccU acc[2][2];
#pragma unroll
    for (int i = 0; i < 2; ++i)
#pragma unroll
        for (int j = 0; j < 2; ++j)
#pragma unroll
            for (int e = 0; e < 8; ++e) acc[i][j].f[e] = 0.0f;

    const uint16_t* Xtile = Xb + (size_t)mb * DIM;
    const uint16_t* Wtile = Wt + ((size_t)z * DIM + nb) * DIM;

    if (wave == 0) {
        tdm_load_2d_bf16(lds_offset_of(&As[0][0][0]), Xtile, 32, 128, DIM, MTOT, DIM);
        tdm_load_2d_bf16(lds_offset_of(&Bs[0][0][0]), Wtile, 32, 64, DIM, 3 * DIM, DIM);
    }

    for (int dk = 0; dk < DIM; dk += 32) {
        const int buf = (dk >> 5) & 1;
        if (wave == 0) __builtin_amdgcn_s_wait_tensorcnt(0);   // current tiles landed
        __syncthreads();                                       // publish; prev buf free
        if (wave == 0 && dk + 32 < DIM) {
            tdm_load_2d_bf16(lds_offset_of(&As[buf ^ 1][0][0]), Xtile + dk + 32,
                             32, 128, DIM, MTOT, DIM);
            tdm_load_2d_bf16(lds_offset_of(&Bs[buf ^ 1][0][0]), Wtile + dk + 32,
                             32, 64, DIM, 3 * DIM, DIM);
        }

        Frag a[2], bfr[2];
#pragma unroll
        for (int mt = 0; mt < 2; ++mt) {
            const uint16_t* ap = &As[buf][wm + mt * 16 + l15][0];
            a[mt].q[0] = *(const uint4*)(ap + aoff);
            a[mt].q[1] = *(const uint4*)(ap + aoff + 8);
        }
#pragma unroll
        for (int nt = 0; nt < 2; ++nt) {
            const uint16_t* bp = &Bs[buf][wn + nt * 16 + l15][0];
            bfr[nt].q[0] = *(const uint4*)(bp + boff);
            bfr[nt].q[1] = *(const uint4*)(bp + boff + 8);
        }
#pragma unroll
        for (int mt = 0; mt < 2; ++mt)
#pragma unroll
            for (int nt = 0; nt < 2; ++nt)
                acc[mt][nt].v = WMMA_BF16(a[mt].v, bfr[nt].v, acc[mt][nt].v);
    }

    // --- epilogue: + bias, bf16 store (V transposed) ---
#pragma unroll
    for (int mt = 0; mt < 2; ++mt) {
#pragma unroll
        for (int nt = 0; nt < 2; ++nt) {
            const int n  = nb + wn + nt * 16 + l15;
            const float bv_ = bias[n];
            const int m0 = mb + wm + mt * 16 + ((lane < 16) ? 0 : 8);
            AccU c = acc[mt][nt];
            if (!transposed) {
#pragma unroll
                for (int v = 0; v < 8; ++v)
                    outp[(size_t)(m0 + v) * DIM + n] = bf16u(c.f[v] + bv_);
            } else {
                uint4 pp;
                pp.x = pk_bf16(c.f[0] + bv_, c.f[1] + bv_);
                pp.y = pk_bf16(c.f[2] + bv_, c.f[3] + bv_);
                pp.z = pk_bf16(c.f[4] + bv_, c.f[5] + bv_);
                pp.w = pk_bf16(c.f[6] + bv_, c.f[7] + bv_);
                *(uint4*)&outp[(size_t)n * MTOT + m0] = pp;  // Vt[d][s]
            }
        }
    }
}

// ---------------------------------------------------------------------------
// Kernel 2: fused sigmoid attention. Block = 32 queries x full D, 8 waves.
// Q tile staged once in LDS (kills the 4x redundant L2 re-reads); K streamed
// from L2-resident bf16; V from transposed Vt (contiguous B-operand loads);
// sigmoid(scores) round-trips through LDS to reach A-operand layout.
// ---------------------------------------------------------------------------
__global__ __launch_bounds__(256) void attn(
    const uint16_t* __restrict__ Qb, const uint16_t* __restrict__ Kb,
    const uint16_t* __restrict__ Vt, float* __restrict__ out)
{
    const int b    = blockIdx.y;
    const int qb   = blockIdx.x * 32;
    const int tid  = threadIdx.x;
    const int wave = tid >> 5;
    const int lane = tid & 31;
    const int mt_w = wave >> 2;      // score-tile query half
    const int nt_w = wave & 3;       // score-tile key subtile
    const int dsl  = wave * 128;     // this wave's D-slice for PV

    __shared__ uint16_t Qs[32][1032]; // 2064B rows: 4-bank rotation/row, 16B aligned
    __shared__ uint16_t Ps[32][72];   // sigmoid(scores) bf16, padded stride

    const int l15  = lane & 15;
    const int aoff = (lane < 16) ? 0 : 16;
    const int boff = (lane < 16) ? 0 : 16;

    const uint16_t* Qrow = Qb + (size_t)(b * SEQ) * DIM;
    const uint16_t* Krow = Kb + (size_t)(b * SEQ) * DIM;

    // --- stage Q tile (32 x 1024 bf16) once ---
    {
        const int r  = tid >> 3;            // 0..31
        const int c0 = (tid & 7) * 128;
        const uint16_t* src = Qrow + (size_t)(qb + r) * DIM + c0;
#pragma unroll
        for (int i = 0; i < 16; ++i)
            *(uint4*)&Qs[r][c0 + i * 8] = *(const uint4*)(src + i * 8);
    }
    __syncthreads();

    AccU acc[2][8];
#pragma unroll
    for (int mt = 0; mt < 2; ++mt)
#pragma unroll
        for (int dt = 0; dt < 8; ++dt)
#pragma unroll
            for (int e = 0; e < 8; ++e) acc[mt][dt].f[e] = 0.0f;

    const uint16_t* Aq = &Qs[mt_w * 16 + l15][0];

    for (int kb = 0; kb < SEQ; kb += 64) {
        // ---- scores: S = Q_tile . K_tile^T over full D ----
        AccU cs;
#pragma unroll
        for (int e = 0; e < 8; ++e) cs.f[e] = 0.0f;

        const uint16_t* Bp = Krow + (size_t)(kb + nt_w * 16 + l15) * DIM;
#pragma unroll 4
        for (int dk = 0; dk < DIM; dk += 32) {
            Frag a, kf;
            a.q[0]  = *(const uint4*)(Aq + dk + aoff);
            a.q[1]  = *(const uint4*)(Aq + dk + aoff + 8);
            kf.q[0] = *(const uint4*)(Bp + dk + boff);
            kf.q[1] = *(const uint4*)(Bp + dk + boff + 8);
            cs.v = WMMA_BF16(a.v, kf.v, cs.v);
        }
        // prefetch next key tile's K rows (gfx1250 global_prefetch_b8)
        if (kb + 64 < SEQ)
            __builtin_prefetch((const void*)(Bp + (size_t)64 * DIM), 0, 1);

        // ---- sigmoid(scale * s) -> bf16 Ps ----
        const float scl = 0.03125f;  // 1/sqrt(1024)
        const int prow = mt_w * 16 + ((lane < 16) ? 0 : 8);
        const int pcol = nt_w * 16 + l15;
#pragma unroll
        for (int v = 0; v < 8; ++v) {
            float p = 1.0f / (1.0f + __expf(-cs.f[v] * scl));
            Ps[prow + v][pcol] = bf16u(p);
        }
        __syncthreads();

        // ---- PV: out_slice += P(32x64) . V(64 x 128-slice) ----
#pragma unroll
        for (int half = 0; half < 2; ++half) {
            const int kk = half * 32;
            Frag pa[2];
#pragma unroll
            for (int mt = 0; mt < 2; ++mt) {
                const uint16_t* pp = &Ps[mt * 16 + l15][kk];
                pa[mt].q[0] = *(const uint4*)(pp + aoff);
                pa[mt].q[1] = *(const uint4*)(pp + aoff + 8);
            }
#pragma unroll
            for (int dt = 0; dt < 8; ++dt) {
                const uint16_t* vp =
                    Vt + (size_t)(dsl + dt * 16 + l15) * MTOT + (b * SEQ + kb + kk + boff);
                Frag vb;
                vb.q[0] = *(const uint4*)vp;
                vb.q[1] = *(const uint4*)(vp + 8);
                acc[0][dt].v = WMMA_BF16(pa[0].v, vb.v, acc[0][dt].v);
                acc[1][dt].v = WMMA_BF16(pa[1].v, vb.v, acc[1][dt].v);
            }
        }
        __syncthreads();
    }

    // ---- epilogue: fp32 output [b][q][d] ----
    float* Op = out + (size_t)(b * SEQ) * DIM;
#pragma unroll
    for (int mt = 0; mt < 2; ++mt) {
        const int m0 = qb + mt * 16 + ((lane < 16) ? 0 : 8);
#pragma unroll
        for (int dt = 0; dt < 8; ++dt) {
            const int n = dsl + dt * 16 + l15;
#pragma unroll
            for (int v = 0; v < 8; ++v)
                Op[(size_t)(m0 + v) * DIM + n] = acc[mt][dt].f[v];
        }
    }
}

// ---------------------------------------------------------------------------
extern "C" void kernel_launch(void* const* d_in, const int* in_sizes, int n_in,
                              void* d_out, int out_size, void* d_ws, size_t ws_size,
                              hipStream_t stream) {
    const float* X  = (const float*)d_in[0];
    const float* Wq = (const float*)d_in[1];
    const float* bq = (const float*)d_in[2];
    const float* Wk = (const float*)d_in[3];
    const float* bk = (const float*)d_in[4];
    const float* Wv = (const float*)d_in[5];
    const float* bv = (const float*)d_in[6];

    // Workspace carve-up (all bf16):
    //   Xb [MTOT][DIM], Wt [3][DIM][DIM], Q [MTOT][DIM], K [MTOT][DIM], Vt [DIM][MTOT]
    uint16_t* Xb = (uint16_t*)d_ws;
    uint16_t* Wt = Xb + (size_t)MTOT * DIM;
    uint16_t* Qw = Wt + (size_t)3 * DIM * DIM;
    uint16_t* Kw = Qw + (size_t)MTOT * DIM;
    uint16_t* Vt = Kw + (size_t)MTOT * DIM;

    cvt_x<<<(size_t)MTOT * DIM / (256 * 8), 256, 0, stream>>>(X, Xb);
    dim3 gW(DIM / 32, DIM / 32, 3);
    cvt_w<<<gW, 256, 0, stream>>>(Wq, Wk, Wv, Wt);

    dim3 gA(DIM / 64, MTOT / 128, 3);
    qkv_gemm<<<gA, 256, 0, stream>>>(Xb, Wt, bq, bk, bv, Qw, Kw, Vt);

    dim3 gB(SEQ / 32, BATCH);
    attn<<<gB, 256, 0, stream>>>(Qw, Kw, Vt, (float*)d_out);
}
